// CausalSelfAttention_4629974745407
// MI455X (gfx1250) — matrix operand order repaired and checked by reading, not compiled, not run on hardware
//
#include <hip/hip_runtime.h>
#include <hip/hip_bf16.h>

typedef __attribute__((ext_vector_type(4)))  __bf16 v4bf;
typedef __attribute__((ext_vector_type(8)))  __bf16 v8bf;
typedef __attribute__((ext_vector_type(16))) __bf16 v16bf;
typedef __attribute__((ext_vector_type(8)))  float  v8f;

#define BB   2
#define TT   2048
#define CC   1024
#define HH   16
#define DD   64
#define MTOT (BB * TT)   // 4096

// ---------- helpers ----------

__device__ __forceinline__ __bf16 f2bf(float f) {
    unsigned u = __builtin_bit_cast(unsigned, f);
    unsigned r = (u + 0x7fffu + ((u >> 16) & 1u)) >> 16;
    unsigned short s = (unsigned short)r;
    return __builtin_bit_cast(__bf16, s);
}

__device__ __forceinline__ v16bf cat8(v8bf lo, v8bf hi) {
    return __builtin_shufflevector(lo, hi, 0,1,2,3,4,5,6,7,8,9,10,11,12,13,14,15);
}

__device__ __forceinline__ v8f wmma_bf16(v16bf a, v16bf b, v8f c) {
    return __builtin_amdgcn_wmma_f32_16x16x32_bf16(
        /*neg_a=*/false, a, /*neg_b=*/false, b,
        /*c_mod=*/(short)0, c, /*reuse_a=*/false, /*reuse_b=*/false);
}

// async memory -> LDS copy, 16 bytes per lane (ASYNCcnt-tracked, no VGPR bounce)
__device__ __forceinline__ void async_copy_b128(unsigned lds_addr, const void* gaddr) {
    asm volatile("global_load_async_to_lds_b128 %0, %1, off"
                 :: "v"(lds_addr), "v"(gaddr) : "memory");
}
__device__ __forceinline__ void wait_async_le1() {
    asm volatile("s_wait_asynccnt 0x1" ::: "memory");
}
__device__ __forceinline__ void wait_async_0() {
    asm volatile("s_wait_asynccnt 0x0" ::: "memory");
}

__device__ __forceinline__ float hmax16(float v) {
    v = fmaxf(v, __shfl_xor(v, 1, 16));
    v = fmaxf(v, __shfl_xor(v, 2, 16));
    v = fmaxf(v, __shfl_xor(v, 4, 16));
    v = fmaxf(v, __shfl_xor(v, 8, 16));
    return v;
}

__device__ __forceinline__ float hsum16(float v) {
    v += __shfl_xor(v, 1, 16);
    v += __shfl_xor(v, 2, 16);
    v += __shfl_xor(v, 4, 16);
    v += __shfl_xor(v, 8, 16);
    return v;
}

// ---------- kernel 1: f32 -> bf16 ----------

__global__ void cvt_bf16_kernel(const float* __restrict__ src,
                                __bf16* __restrict__ dst, int n) {
    int i = (blockIdx.x * 256 + threadIdx.x) * 4;
    if (i >= n) return;
    float4 v = *(const float4*)(src + i);
    v4bf o;
    o.x = f2bf(v.x); o.y = f2bf(v.y); o.z = f2bf(v.z); o.w = f2bf(v.w);
    *(v4bf*)(dst + i) = o;
}

// ---------- kernel 2: QKV GEMM + bias + scatter to Q / K^T / V (bf16) ----------
// A: [MTOT, CC] bf16 (x), B: [CC, 3*CC] bf16 (w_attn).
// Block: 8 waves, 256x64 tile (32 rows/wave); B tile double-buffered via async->LDS.

__global__ __launch_bounds__(256)
void gemm_qkv_kernel(const __bf16* __restrict__ A, const __bf16* __restrict__ Bm,
                     const float* __restrict__ bias,
                     __bf16* __restrict__ Qb, __bf16* __restrict__ Ktb,
                     __bf16* __restrict__ Vb) {
    const int NQ = 3 * CC;
    __shared__ __attribute__((aligned(32))) __bf16 lds_b[2][32 * 64];
    const int tid  = threadIdx.x;
    const int wave = tid >> 5;
    const int lane = tid & 31;
    const int l16  = lane & 15;
    const int hb   = lane >> 4;
    const int m0   = blockIdx.y * 256 + wave * 32;
    const int n0   = blockIdx.x * 64;

    // this thread's 16-byte slice of the 32x64 B tile
    const int brow = tid >> 3;
    const int bcol = (tid & 7) * 8;
    const __bf16* gsrc = Bm + (size_t)brow * NQ + n0 + bcol;
    const unsigned ldst0 = (unsigned)(size_t)(&lds_b[0][brow * 64 + bcol]);
    const unsigned ldst1 = (unsigned)(size_t)(&lds_b[1][brow * 64 + bcol]);

    v8f acc[2][4] = {};
    const int NK = CC / 32;

    async_copy_b128(ldst0, gsrc);                       // prologue: tile 0
    for (int j = 0; j < NK; ++j) {
        if (j + 1 < NK) {
            async_copy_b128((j & 1) ? ldst0 : ldst1,
                            gsrc + (size_t)(j + 1) * 32 * NQ);
            wait_async_le1();                           // tile j landed
        } else {
            wait_async_0();
        }
        __syncthreads();                                // all waves' slices landed

        const __bf16* cur = lds_b[j & 1];
        const int k0 = j * 32;
        const __bf16* arow0 = A + (size_t)(m0 + l16) * CC + k0 + hb * 8;
        const __bf16* arow1 = arow0 + (size_t)16 * CC;
        v16bf af0 = cat8(*(const v8bf*)(arow0), *(const v8bf*)(arow0 + 16));
        v16bf af1 = cat8(*(const v8bf*)(arow1), *(const v8bf*)(arow1 + 16));

        #pragma unroll
        for (int f = 0; f < 4; ++f) {
            v16bf bf;
            for (int s = 0; s < 16; ++s) bf[s] = cur[(hb * 8 + (s & 7) + (s >> 3) * 16) * 64 + f * 16 + l16];
            acc[0][f] = wmma_bf16(af0, bf, acc[0][f]);
            acc[1][f] = wmma_bf16(af1, bf, acc[1][f]);
        }
        __syncthreads();                                // reads done before buffer reuse
    }

    // epilogue: bias + scatter into [B,H,T,D] (Q,V) and [B,H,D,T] (K^T), bf16
    #pragma unroll
    for (int mi = 0; mi < 2; ++mi) {
        #pragma unroll
        for (int f = 0; f < 4; ++f) {
            int n   = n0 + f * 16 + l16;
            float bv = bias[n];
            int sec = n >> 10;       // 0=Q 1=K 2=V
            int c   = n & 1023;
            int h   = c >> 6;
            int d   = c & 63;
            #pragma unroll
            for (int r = 0; r < 8; ++r) {
                int m = m0 + mi * 16 + hb * 8 + r;
                int b = m >> 11;
                int t = m & 2047;
                int bh = b * HH + h;
                __bf16 val = f2bf(acc[mi][f][r] + bv);
                if (sec == 0)      Qb [((size_t)(bh * TT + t)) * DD + d] = val;
                else if (sec == 1) Ktb[((size_t)(bh * DD + d)) * TT + t] = val;
                else               Vb [((size_t)(bh * TT + t)) * DD + d] = val;
            }
        }
    }
}

// ---------- kernel 3: causal flash attention (1 wave per 16-query tile) ----------
// O written as bf16 in [B,T,C] layout, ready for the projection GEMM.

__global__ __launch_bounds__(32)
void flash_attn_kernel(const __bf16* __restrict__ Qb, const __bf16* __restrict__ Ktb,
                       const __bf16* __restrict__ Vb, __bf16* __restrict__ Ob) {
    __shared__ __attribute__((aligned(32))) __bf16 lds_p[16 * 32];
    const int lane = threadIdx.x;
    const int l16  = lane & 15;
    const int hb   = lane >> 4;
    const int qt   = blockIdx.x;
    const int bh   = blockIdx.y;
    const int b    = bh >> 4;
    const int h    = bh & 15;
    const int q0   = qt * 16;

    const __bf16* qbase  = Qb  + (size_t)bh * TT * DD;
    const __bf16* ktbase = Ktb + (size_t)bh * DD * TT;
    const __bf16* vbase  = Vb  + (size_t)bh * TT * DD;

    // Q A-fragments: D=64 contraction = two K=32 chunks
    const __bf16* qrow = qbase + (size_t)(q0 + l16) * DD + hb * 8;
    v16bf qa0 = cat8(*(const v8bf*)(qrow),      *(const v8bf*)(qrow + 16));
    v16bf qa1 = cat8(*(const v8bf*)(qrow + 32), *(const v8bf*)(qrow + 48));

    v8f o[4] = {};
    float rowmax[8], rowsum[8];
    #pragma unroll
    for (int r = 0; r < 8; ++r) { rowmax[r] = -3.0e38f; rowsum[r] = 0.f; }

    const int nblk   = (q0 + 47) >> 5;     // 32-key blocks covering keys 0..q0+15
    const float scale = 0.125f;            // 1/sqrt(64)

    for (int j = 0; j < nblk; ++j) {
        // S = Q K^T : two 16x16 fragments, K^T rows are contiguous (lane = key)
        v8f s0 = {}, s1 = {};
        {
            v16bf kb00, kb01, kb10, kb11;
            for (int s = 0; s < 16; ++s) {
                int kd = hb * 8 + (s & 7) + (s >> 3) * 16;
                kb00[s] = ktbase[(size_t)(kd)      * TT + j * 32 + l16];
                kb01[s] = ktbase[(size_t)(32 + kd) * TT + j * 32 + l16];
                kb10[s] = ktbase[(size_t)(kd)      * TT + j * 32 + 16 + l16];
                kb11[s] = ktbase[(size_t)(32 + kd) * TT + j * 32 + 16 + l16];
            }
            s0 = wmma_bf16(qa0, kb00, s0);
            s0 = wmma_bf16(qa1, kb01, s0);
            s1 = wmma_bf16(qa0, kb10, s1);
            s1 = wmma_bf16(qa1, kb11, s1);
        }

        // online softmax (rows in VGPR index, columns across 16-lane halves)
        #pragma unroll
        for (int r = 0; r < 8; ++r) {
            int q   = q0 + hb * 8 + r;
            int k0i = j * 32 + l16;
            float v0 = (k0i      <= q) ? s0[r] * scale : -3.0e38f;
            float v1 = (k0i + 16 <= q) ? s1[r] * scale : -3.0e38f;
            float rm = hmax16(fmaxf(v0, v1));
            float mn = fmaxf(rowmax[r], rm);
            float alpha = __expf(rowmax[r] - mn);
            rowmax[r] = mn;
            float p0 = __expf(v0 - mn);
            float p1 = __expf(v1 - mn);
            float rs = hsum16(p0 + p1);
            rowsum[r] = rowsum[r] * alpha + rs;
            #pragma unroll
            for (int dn = 0; dn < 4; ++dn) o[dn][r] *= alpha;
            lds_p[(hb * 8 + r) * 32 + l16]      = f2bf(p0);
            lds_p[(hb * 8 + r) * 32 + 16 + l16] = f2bf(p1);
        }
        __syncthreads();

        // O += P V : P re-swizzled C->A layout via LDS, V rows contiguous (lane = d)
        const __bf16* prow = lds_p + l16 * 32 + hb * 8;
        v16bf pa = cat8(*(const v8bf*)(prow), *(const v8bf*)(prow + 16));
        const __bf16* vrow = vbase + (size_t)(j * 32) * DD + l16;
        #pragma unroll
        for (int dn = 0; dn < 4; ++dn) {
            v16bf vb;
            for (int s = 0; s < 16; ++s) vb[s] = vrow[(size_t)(hb * 8 + (s & 7) + (s >> 3) * 16) * DD + dn * 16];
            o[dn] = wmma_bf16(pa, vb, o[dn]);
        }
        __syncthreads();
    }

    #pragma unroll
    for (int r = 0; r < 8; ++r) rowsum[r] = 1.0f / rowsum[r];
    #pragma unroll
    for (int dn = 0; dn < 4; ++dn) {
        #pragma unroll
        for (int r = 0; r < 8; ++r) {
            size_t idx = ((size_t)(b * TT + q0 + hb * 8 + r)) * CC + h * DD + dn * 16 + l16;
            Ob[idx] = f2bf(o[dn][r] * rowsum[r]);
        }
    }
}

// ---------- kernel 4: output projection GEMM + bias (f32 out) ----------

__global__ __launch_bounds__(256)
void gemm_proj_kernel(const __bf16* __restrict__ A, const __bf16* __restrict__ Bm,
                      const float* __restrict__ bias, float* __restrict__ out) {
    const int N = CC;
    __shared__ __attribute__((aligned(32))) __bf16 lds_b[2][32 * 64];
    const int tid  = threadIdx.x;
    const int wave = tid >> 5;
    const int lane = tid & 31;
    const int l16  = lane & 15;
    const int hb   = lane >> 4;
    const int m0   = blockIdx.y * 256 + wave * 32;
    const int n0   = blockIdx.x * 64;

    const int brow = tid >> 3;
    const int bcol = (tid & 7) * 8;
    const __bf16* gsrc = Bm + (size_t)brow * N + n0 + bcol;
    const unsigned ldst0 = (unsigned)(size_t)(&lds_b[0][brow * 64 + bcol]);
    const unsigned ldst1 = (unsigned)(size_t)(&lds_b[1][brow * 64 + bcol]);

    v8f acc[2][4] = {};
    const int NK = CC / 32;

    async_copy_b128(ldst0, gsrc);
    for (int j = 0; j < NK; ++j) {
        if (j + 1 < NK) {
            async_copy_b128((j & 1) ? ldst0 : ldst1,
                            gsrc + (size_t)(j + 1) * 32 * N);
            wait_async_le1();
        } else {
            wait_async_0();
        }
        __syncthreads();

        const __bf16* cur = lds_b[j & 1];
        const int k0 = j * 32;
        const __bf16* arow0 = A + (size_t)(m0 + l16) * CC + k0 + hb * 8;
        const __bf16* arow1 = arow0 + (size_t)16 * CC;
        v16bf af0 = cat8(*(const v8bf*)(arow0), *(const v8bf*)(arow0 + 16));
        v16bf af1 = cat8(*(const v8bf*)(arow1), *(const v8bf*)(arow1 + 16));

        #pragma unroll
        for (int f = 0; f < 4; ++f) {
            v16bf bf;
            for (int s = 0; s < 16; ++s) bf[s] = cur[(hb * 8 + (s & 7) + (s >> 3) * 16) * 64 + f * 16 + l16];
            acc[0][f] = wmma_bf16(af0, bf, acc[0][f]);
            acc[1][f] = wmma_bf16(af1, bf, acc[1][f]);
        }
        __syncthreads();
    }

    #pragma unroll
    for (int mi = 0; mi < 2; ++mi) {
        #pragma unroll
        for (int f = 0; f < 4; ++f) {
            int n = n0 + f * 16 + l16;
            float bv = bias[n];
            #pragma unroll
            for (int r = 0; r < 8; ++r) {
                int m = m0 + mi * 16 + hb * 8 + r;
                out[(size_t)m * N + n] = acc[mi][f][r] + bv;
            }
        }
    }
}

// ---------- launch ----------

extern "C" void kernel_launch(void* const* d_in, const int* in_sizes, int n_in,
                              void* d_out, int out_size, void* d_ws, size_t ws_size,
                              hipStream_t stream) {
    const float* x      = (const float*)d_in[0];
    const float* w_attn = (const float*)d_in[1];
    const float* b_attn = (const float*)d_in[2];
    const float* w_proj = (const float*)d_in[3];
    const float* b_proj = (const float*)d_in[4];
    float* out = (float*)d_out;

    char* ws = (char*)d_ws;
    size_t off = 0;
    auto take = [&](size_t elems) {
        void* p = ws + off;
        off += (elems * sizeof(__bf16) + 255) & ~(size_t)255;
        return (__bf16*)p;
    };

    const int nX  = MTOT * CC;         // 4,194,304
    const int nWa = CC * 3 * CC;       // 3,145,728
    const int nWp = CC * CC;           // 1,048,576
    const int nHD = BB * HH * TT * DD; // 4,194,304

    __bf16* Xb  = take(nX);
    __bf16* Wab = take(nWa);
    __bf16* Wpb = take(nWp);
    __bf16* Qb  = take(nHD);
    __bf16* Ktb = take(nHD);
    __bf16* Vb  = take(nHD);
    __bf16* Ob  = take(nX);

    // 1) convert inputs to bf16
    cvt_bf16_kernel<<<(nX  / 4 + 255) / 256, 256, 0, stream>>>(x,      Xb,  nX);
    cvt_bf16_kernel<<<(nWa / 4 + 255) / 256, 256, 0, stream>>>(w_attn, Wab, nWa);
    cvt_bf16_kernel<<<(nWp / 4 + 255) / 256, 256, 0, stream>>>(w_proj, Wpb, nWp);

    // 2) QKV projection + scatter (256x64 block tile, async double-buffered B)
    gemm_qkv_kernel<<<dim3(3 * CC / 64, MTOT / 256), 256, 0, stream>>>(
        Xb, Wab, b_attn, Qb, Ktb, Vb);

    // 3) causal flash attention
    flash_attn_kernel<<<dim3(TT / 16, BB * HH), 32, 0, stream>>>(Qb, Ktb, Vb, Ob);

    // 4) output projection
    gemm_proj_kernel<<<dim3(CC / 64, MTOT / 256), 256, 0, stream>>>(
        Ob, Wpb, b_proj, out);
}
